// GatedLinearAttention_54494545051923
// MI455X (gfx1250) — matrix operand order repaired and checked by reading, not compiled, not run on hardware
//
#include <hip/hip_runtime.h>

// ---------------- problem constants ----------------
constexpr int kB    = 2;
constexpr int kS    = 4096;
constexpr int kH    = 8;
constexpr int kDK   = 128;
constexpr int kDV   = 256;
constexpr int kDim  = 1024;
constexpr int kHDK  = kH * kDK;   // 1024
constexpr int kHDV  = kH * kDV;   // 2048
constexpr int kM    = kB * kS;    // 8192 rows

typedef __attribute__((ext_vector_type(16))) _Float16 v16h;
typedef __attribute__((ext_vector_type(8)))  _Float16 v8h;
typedef __attribute__((ext_vector_type(8)))  float    v8f;

__device__ inline v8f zero8() {
    v8f z;
#pragma unroll
    for (int i = 0; i < 8; ++i) z[i] = 0.f;
    return z;
}

__device__ inline v8f wmma16(v16h a, v16h b, v8f c) {
    return __builtin_amdgcn_wmma_f32_16x16x32_f16(false, a, false, b, (short)0, c,
                                                  false, false);
}

__device__ inline v16h combine16(v8h lo, v8h hi) {
    v16h a;
#pragma unroll
    for (int i = 0; i < 8; ++i) { a[i] = lo[i]; a[8 + i] = hi[i]; }
    return a;
}

// ---- CDNA5 async copy: global -> LDS, 16B per lane, tracked by ASYNCcnt ----
__device__ inline void async_b128(void* lds, const void* g) {
    uint32_t la = (uint32_t)(uintptr_t)lds;     // LDS offset = low 32 bits
    uint64_t ga = (uint64_t)(uintptr_t)g;
    asm volatile("global_load_async_to_lds_b128 %0, %1, off"
                 :: "v"(la), "v"(ga) : "memory");
}
__device__ inline void wait_async0() {
    asm volatile("s_wait_asynccnt 0x0" ::: "memory");
}

// A fragment (16x32 f16): lanes 0-15 row M=lane, halves K=[kh*8,+8) and [16+kh*8,+8)
__device__ inline v16h load_fragA(const _Float16* lds, int stride, int row0, int kk0, int lane) {
    const _Float16* p = lds + (row0 + (lane & 15)) * stride + kk0 + ((lane >> 4) << 3);
    return combine16(*(const v8h*)p, *(const v8h*)(p + 16));
}
// B fragment from B^T layout [n][k]: lane n=lane&15, halves K=[kh*8,+8) and [16+kh*8,+8)
__device__ inline v16h load_fragB(const _Float16* lds, int stride, int n0, int kk0, int lane) {
    const _Float16* p = lds + (n0 + (lane & 15)) * stride + kk0 + ((lane >> 4) << 3);
    return combine16(*(const v8h*)p, *(const v8h*)(p + 16));
}
// A fragment straight from global row-major f16 [row][K]
__device__ inline v16h load_fragA_g(const _Float16* g, int ldg, int row0, int kk0, int lane) {
    const _Float16* p = g + (size_t)(row0 + (lane & 15)) * ldg + kk0 + ((lane >> 4) << 3);
    return combine16(*(const v8h*)p, *(const v8h*)(p + 16));
}
// B fragment straight from global row-major f16 [n][K]
__device__ inline v16h load_fragB_g(const _Float16* g, int ldg, int n0, int kk0, int lane) {
    const _Float16* p = g + (size_t)(n0 + (lane & 15)) * ldg + kk0 + ((lane >> 4) << 3);
    return combine16(*(const v8h*)p, *(const v8h*)(p + 16));
}

// ---------------- prep: f32 -> f16 ----------------
__global__ __launch_bounds__(256) void cvt_f16_kernel(const float* __restrict__ in,
                                                      _Float16* __restrict__ out, int n) {
    int i = (blockIdx.x * 256 + threadIdx.x) * 4;
    if (i < n) {
        float4 f = *(const float4*)(in + i);
        out[i + 0] = (_Float16)f.x; out[i + 1] = (_Float16)f.y;
        out[i + 2] = (_Float16)f.z; out[i + 3] = (_Float16)f.w;
    }
}

// ---------------- prep: W f32 [K,N] -> W^T f16 [N,K] (scale folded) ----------
__global__ __launch_bounds__(256) void wtrans_kernel(const float* __restrict__ W,
                                                     _Float16* __restrict__ WT,
                                                     int K, int N, float scale) {
    __shared__ float tile[32][33];
    const int bk = blockIdx.y * 32, bn = blockIdx.x * 32;
    const int tx = threadIdx.x & 31, ty = threadIdx.x >> 5;
    for (int r = ty; r < 32; r += 8)
        tile[r][tx] = W[(size_t)(bk + r) * N + bn + tx];
    __syncthreads();
    for (int r = ty; r < 32; r += 8)
        WT[(size_t)(bn + r) * K + bk + tx] = (_Float16)(tile[tx][r] * scale);
}

// -------- unified async double-buffered GEMM: C = epi(A_f16[M,K] @ BT_f16[N,K]^T)
constexpr int BM = 128, BN = 128, BK = 32, LSTR = 40;

__device__ inline void stage_tile_async(const _Float16* A, const _Float16* BT,
                                        _Float16* sAb, _Float16* sBb,
                                        int m0, int n0, int k0, int Kdim, int tid) {
#pragma unroll
    for (int t = 0; t < 2; ++t) {
        int idx = tid + t * 256;           // 512 chunk-ids: row=idx>>2, chunk=idx&3
        int row = idx >> 2, c8 = (idx & 3) << 3;
        async_b128(sAb + row * LSTR + c8, A + (size_t)(m0 + row) * Kdim + k0 + c8);
        async_b128(sBb + row * LSTR + c8, BT + (size_t)(n0 + row) * Kdim + k0 + c8);
    }
}

template <int EPI>  // 0: ->f16, 2: sigmoid ->f16, 3: ->f32
__global__ __launch_bounds__(256) void gemm_f16(const _Float16* __restrict__ A,
                                                const _Float16* __restrict__ BT,
                                                void* __restrict__ Cout,
                                                int Kdim, int Ndim) {
    __shared__ alignas(16) _Float16 sA[2][BM * LSTR];
    __shared__ alignas(16) _Float16 sB[2][BN * LSTR];
    const int tid = threadIdx.x, lane = tid & 31, wv = tid >> 5;
    const int m0 = blockIdx.y * BM, n0 = blockIdx.x * BN;
    const int wm = (wv & 3) * 32, wn = (wv >> 2) * 64;

    v8f acc[2][4];
#pragma unroll
    for (int i = 0; i < 2; ++i)
#pragma unroll
        for (int j = 0; j < 4; ++j) acc[i][j] = zero8();

    const int nk = Kdim / BK;
    stage_tile_async(A, BT, sA[0], sB[0], m0, n0, 0, Kdim, tid);
    wait_async0();
    __syncthreads();

    for (int kt = 0; kt < nk; ++kt) {
        const int cur = kt & 1;
        if (kt + 1 < nk)   // DMA next tile while computing this one
            stage_tile_async(A, BT, sA[1 - cur], sB[1 - cur], m0, n0,
                             (kt + 1) * BK, Kdim, tid);
        v16h af[2], bf[4];
#pragma unroll
        for (int i = 0; i < 2; ++i) af[i] = load_fragA(sA[cur], LSTR, wm + i * 16, 0, lane);
#pragma unroll
        for (int j = 0; j < 4; ++j) bf[j] = load_fragB(sB[cur], LSTR, wn + j * 16, 0, lane);
#pragma unroll
        for (int i = 0; i < 2; ++i)
#pragma unroll
            for (int j = 0; j < 4; ++j) acc[i][j] = wmma16(af[i], bf[j], acc[i][j]);
        wait_async0();       // next tile landed
        __syncthreads();     // everyone done reading cur
    }

    const int cb = n0 + wn;
#pragma unroll
    for (int i = 0; i < 2; ++i) {
        int rb = m0 + wm + i * 16 + ((lane >> 4) << 3);
#pragma unroll
        for (int j = 0; j < 4; ++j) {
            int col = cb + j * 16 + (lane & 15);
#pragma unroll
            for (int r = 0; r < 8; ++r) {
                float v = acc[i][j][r];
                if (EPI == 2) v = 1.f / (1.f + __expf(-v));
                if (EPI == 3)
                    ((float*)Cout)[(size_t)(rb + r) * Ndim + col] = v;
                else
                    ((_Float16*)Cout)[(size_t)(rb + r) * Ndim + col] = (_Float16)v;
            }
        }
    }
}

// ---------------- alpha = sigmoid((x@Wg + bg)/16) ------------------------------
__global__ __launch_bounds__(256) void alpha_kernel(const float* __restrict__ x,
                                                    const float* __restrict__ Wg,
                                                    const float* __restrict__ bg,
                                                    float* __restrict__ alpha) {
    const int lane = threadIdx.x & 31;
    const int h = threadIdx.x >> 5;
    const int r = blockIdx.x;
    float sum = 0.f;
    for (int i = lane; i < kDim; i += 32)
        sum += x[(size_t)r * kDim + i] * Wg[i * kH + h];
#pragma unroll
    for (int m = 16; m; m >>= 1) sum += __shfl_xor(sum, m, 32);
    if (lane == 0)
        alpha[(size_t)r * kH + h] = 1.f / (1.f + __expf(-(sum + bg[h]) * 0.0625f));
}

// ---------------- gate: ogh <- oh * ogh (recomputed every call) ----------------
__global__ __launch_bounds__(256) void gate_kernel(const _Float16* __restrict__ oh,
                                                   _Float16* __restrict__ ogh) {
    size_t i = ((size_t)blockIdx.x * 256 + threadIdx.x) * 8;
    v8h o = *(const v8h*)(oh + i);
    v8h g = *(const v8h*)(ogh + i);
    v8h r;
#pragma unroll
    for (int e = 0; e < 8; ++e) r[e] = (_Float16)((float)o[e] * (float)g[e]);
    *(v8h*)(ogh + i) = r;
}

// ---------------- chunked GLA scan --------------------------------------------
constexpr int CCH = 64;
constexpr int DVB = 64;
constexpr int SKT = 72;    // sKT row stride (halfs, mult of 8)
constexpr int SVS = 72;    // sVT / sD row stride
constexpr int SST = 136;   // sShT row stride (dk=128 + pad)

__global__ __launch_bounds__(256) void gla_kernel(const _Float16* __restrict__ qh,
                                                  const _Float16* __restrict__ kh,
                                                  const _Float16* __restrict__ vh,
                                                  const float* __restrict__ alpha,
                                                  _Float16* __restrict__ oh) {
    __shared__ alignas(16) _Float16 sKT [kDK * SKT];   // (r_s * k)^T  [dk][s]
    __shared__ alignas(16) _Float16 sVT [DVB * SVS];   // v^T          [dv][s]
    __shared__ alignas(16) _Float16 sShT[DVB * SST];   // state^T f16  [dv][dk]
    __shared__ alignas(16) _Float16 sD  [CCH * SVS];   // decayed masked scores [t][s]
    __shared__ float sL[CCH];

    const int tid = threadIdx.x, lane = tid & 31, wv = tid >> 5;
    const int nVB = kDV / DVB;
    const int bh = blockIdx.x / nVB, vb = blockIdx.x % nVB;
    const int b = bh / kH, h = bh % kH;

    const _Float16* qb = qh + (size_t)b * kS * kHDK + h * kDK;
    const _Float16* kb = kh + (size_t)b * kS * kHDK + h * kDK;
    const _Float16* vp = vh + (size_t)b * kS * kHDV + h * kDV + vb * DVB;
    _Float16*       ob = oh + (size_t)b * kS * kHDV + h * kDV + vb * DVB;
    const float*    ab = alpha + (size_t)b * kS * kH + h;

    v8f vstate[4];
#pragma unroll
    for (int t = 0; t < 4; ++t) vstate[t] = zero8();

    const int tm  = (wv & 3) * 16;
    const int tn0 = (wv >> 2) * 32;

    for (int ch = 0; ch < kS / CCH; ++ch) {
        const int s0 = ch * CCH;
        __syncthreads();
        if (tid == 0) {
            float a = 0.f;
            for (int t = 0; t < CCH; ++t) { a += __logf(ab[(size_t)(s0 + t) * kH]); sL[t] = a; }
        }
        // prefetch next chunk (global_prefetch_b8)
        if (ch + 1 < kS / CCH && tid < CCH) {
            const int sp = s0 + CCH + tid;
            __builtin_prefetch(qb + (size_t)sp * kHDK, 0, 3);
            __builtin_prefetch(kb + (size_t)sp * kHDK, 0, 3);
            __builtin_prefetch(vp + (size_t)sp * kHDV, 0, 3);
        }
        // stage v^T
        for (int i = tid; i < CCH * DVB / 8; i += 256) {
            int t = i >> 3, c8 = (i & 7) << 3;
            v8h vv = *(const v8h*)(vp + (size_t)(s0 + t) * kHDV + c8);
#pragma unroll
            for (int e = 0; e < 8; ++e) sVT[(c8 + e) * SVS + t] = vv[e];
        }
        // state^T f16 snapshot: one contiguous v8h store per tile per lane
        {
            int dkb = wv * 16 + ((lane >> 4) << 3);
#pragma unroll
            for (int t = 0; t < 4; ++t) {
                int dv = t * 16 + (lane & 15);
                v8h s;
#pragma unroll
                for (int r = 0; r < 8; ++r) s[r] = (_Float16)vstate[t][r];
                *(v8h*)(sShT + dv * SST + dkb) = s;
            }
        }
        __syncthreads();
        const float LC = sL[CCH - 1];
        // stage (r_s * k)^T
        for (int i = tid; i < CCH * kDK / 8; i += 256) {
            int t = i >> 4, c8 = (i & 15) << 3;
            float rs = __expf(LC - sL[t]);
            v8h kv = *(const v8h*)(kb + (size_t)(s0 + t) * kHDK + c8);
#pragma unroll
            for (int e = 0; e < 8; ++e)
                sKT[(c8 + e) * SKT + t] = (_Float16)((float)kv[e] * rs);
        }
        // q fragments (raw + P_t-scaled)
        v16h aq[4], aqp[4];
        {
            float pt = __expf(sL[tm + (lane & 15)]);
#pragma unroll
            for (int kt = 0; kt < 4; ++kt) {
                aq[kt] = load_fragA_g(qb, kHDK, s0 + tm, kt * 32, lane);
                v16h s;
#pragma unroll
                for (int e = 0; e < 16; ++e) s[e] = (_Float16)((float)aq[kt][e] * pt);
                aqp[kt] = s;
            }
        }
        // WMMA1: scores + decay + causal mask -> sD
#pragma unroll
        for (int j = 0; j < 2; ++j) {
            v8f acc = zero8();
#pragma unroll
            for (int kt = 0; kt < 4; ++kt) {
                v16h bf = load_fragB_g(kb, kHDK, s0 + tn0 + j * 16, kt * 32, lane);
                acc = wmma16(aq[kt], bf, acc);
            }
            int scol = tn0 + j * 16 + (lane & 15);
            float ls = sL[scol];
            int tb = tm + ((lane >> 4) << 3);
#pragma unroll
            for (int r = 0; r < 8; ++r) {
                int trow = tb + r;
                float v = (trow >= scol) ? acc[r] * __expf(sL[trow] - ls) : 0.f;
                sD[trow * SVS + scol] = (_Float16)v;
            }
        }
        __syncthreads();
        // WMMA2: O = (P_t q) @ S0 + D @ v
#pragma unroll
        for (int j = 0; j < 2; ++j) {
            v8f acc = zero8();
#pragma unroll
            for (int kt = 0; kt < 4; ++kt) {
                v16h bf = load_fragB(sShT, SST, tn0 + j * 16, kt * 32, lane);
                acc = wmma16(aqp[kt], bf, acc);
            }
#pragma unroll
            for (int kt = 0; kt < 2; ++kt) {
                v16h af = load_fragA(sD, SVS, tm, kt * 32, lane);
                v16h bf = load_fragB(sVT, SVS, tn0 + j * 16, kt * 32, lane);
                acc = wmma16(af, bf, acc);
            }
            int dv = tn0 + j * 16 + (lane & 15);
            int tb = tm + ((lane >> 4) << 3);
#pragma unroll
            for (int r = 0; r < 8; ++r)
                ob[(size_t)(s0 + tb + r) * kHDV + dv] = (_Float16)acc[r];
        }
        // WMMA3: S = P_C * S + (r_s k)^T @ v
        {
            const float pc = __expf(LC);
#pragma unroll
            for (int t = 0; t < 4; ++t) {
                v8f acc = vstate[t];
#pragma unroll
                for (int e = 0; e < 8; ++e) acc[e] *= pc;
#pragma unroll
                for (int kt = 0; kt < 2; ++kt) {
                    v16h af = load_fragA(sKT, SKT, wv * 16, kt * 32, lane);
                    v16h bf = load_fragB(sVT, SVS, t * 16, kt * 32, lane);
                    acc = wmma16(af, bf, acc);
                }
                vstate[t] = acc;
            }
        }
    }
}

// ---------------- launcher -----------------------------------------------------
extern "C" void kernel_launch(void* const* d_in, const int* in_sizes, int n_in,
                              void* d_out, int out_size, void* d_ws, size_t ws_size,
                              hipStream_t stream) {
    const float* x   = (const float*)d_in[0];
    const float* Wq  = (const float*)d_in[1];
    const float* Wk  = (const float*)d_in[2];
    const float* Wv  = (const float*)d_in[3];
    const float* Wo  = (const float*)d_in[4];
    const float* Wg  = (const float*)d_in[5];
    const float* bg  = (const float*)d_in[6];
    const float* Wog = (const float*)d_in[7];

    char* ws = (char*)d_ws;
    _Float16* qh   = (_Float16*)ws;  ws += (size_t)kM * kHDK * 2;     // 16.8 MB
    _Float16* khp  = (_Float16*)ws;  ws += (size_t)kM * kHDK * 2;     // 16.8 MB
    _Float16* vhp  = (_Float16*)ws;  ws += (size_t)kM * kHDV * 2;     // 33.6 MB
    _Float16* ogh  = (_Float16*)ws;  ws += (size_t)kM * kHDV * 2;     // 33.6 MB
    _Float16* ohp  = (_Float16*)ws;  ws += (size_t)kM * kHDV * 2;     // 33.6 MB
    float*    alp  = (float*)ws;     ws += (size_t)kM * kH * 4;       // 0.26 MB
    _Float16* xh   = (_Float16*)ws;  ws += (size_t)kM * kDim * 2;     // 16.8 MB
    _Float16* WqT  = (_Float16*)ws;  ws += (size_t)kHDK * kDim * 2;   // 2 MB
    _Float16* WkT  = (_Float16*)ws;  ws += (size_t)kHDK * kDim * 2;   // 2 MB
    _Float16* WvT  = (_Float16*)ws;  ws += (size_t)kHDV * kDim * 2;   // 4 MB
    _Float16* WogT = (_Float16*)ws;  ws += (size_t)kHDV * kDim * 2;   // 4 MB
    _Float16* WoT  = (_Float16*)ws;  ws += (size_t)kDim * kHDV * 2;   // 4 MB

    dim3 blk(256);
    // prep: f16 conversions + weight transposes (DK^-0.5 folded into WkT)
    cvt_f16_kernel<<<dim3((kM * kDim) / 1024), blk, 0, stream>>>(x, xh, kM * kDim);
    wtrans_kernel<<<dim3(kHDK / 32, kDim / 32), blk, 0, stream>>>(Wq,  WqT,  kDim, kHDK, 1.f);
    wtrans_kernel<<<dim3(kHDK / 32, kDim / 32), blk, 0, stream>>>(Wk,  WkT,  kDim, kHDK, 0.08838834764831844f);
    wtrans_kernel<<<dim3(kHDV / 32, kDim / 32), blk, 0, stream>>>(Wv,  WvT,  kDim, kHDV, 1.f);
    wtrans_kernel<<<dim3(kHDV / 32, kDim / 32), blk, 0, stream>>>(Wog, WogT, kDim, kHDV, 1.f);
    wtrans_kernel<<<dim3(kDim / 32, kHDV / 32), blk, 0, stream>>>(Wo,  WoT,  kHDV, kDim, 1.f);
    // projections (async double-buffered WMMA GEMMs)
    gemm_f16<0><<<dim3(kHDK / BN, kM / BM), blk, 0, stream>>>(xh, WqT,  qh,  kDim, kHDK);
    gemm_f16<0><<<dim3(kHDK / BN, kM / BM), blk, 0, stream>>>(xh, WkT,  khp, kDim, kHDK);
    gemm_f16<0><<<dim3(kHDV / BN, kM / BM), blk, 0, stream>>>(xh, WvT,  vhp, kDim, kHDV);
    gemm_f16<2><<<dim3(kHDV / BN, kM / BM), blk, 0, stream>>>(xh, WogT, ogh, kDim, kHDV);
    alpha_kernel<<<dim3(kM), blk, 0, stream>>>(x, Wg, bg, alp);
    // chunked scan
    gla_kernel<<<dim3(kB * kH * (kDV / DVB)), blk, 0, stream>>>(qh, khp, vhp, alp, ohp);
    // gate + output projection
    gate_kernel<<<dim3((kM * kHDV) / (256 * 8)), blk, 0, stream>>>(ohp, ogh);
    gemm_f16<3><<<dim3(kDim / BN, kM / BM), blk, 0, stream>>>(ogh, WoT, (float*)d_out, kHDV, kDim);
}
